// DRL4TSP_27324581937281
// MI455X (gfx1250) — matrix operand order repaired and checked by reading, not compiled, not run on hardware
//
#include <hip/hip_runtime.h>
#include <hip/hip_bf16.h>
#include <stdint.h>

// DRL4TSP pointer-network decode for MI455X (gfx1250, wave32, WMMA).
// B=2048, H=128, S=128.  Dominant cost: 4.3G tanh + per-step re-read of
// base(B,H,S) fp32 = 134MB (kept L2-resident; 192MB global L2).
// LDS weight buffers are stored in WMMA *fragment order* so every B-operand
// is two aligned ds_load_b128 (no strided gathers / register shuffling).

#define H 128
#define S 128
#define BATCH 2048
#define BTILE 16
#define NTHREADS 256

typedef __attribute__((ext_vector_type(16))) __bf16   v16bf;
typedef __attribute__((ext_vector_type(8)))  float    v8f;
typedef __attribute__((ext_vector_type(8)))  uint32_t v8u;

static __device__ __forceinline__ uint16_t f2bf(float f) {
    uint32_t u = __builtin_bit_cast(uint32_t, f);
    u += 0x7FFFu + ((u >> 16) & 1u);          // round-to-nearest-even
    return (uint16_t)(u >> 16);
}

#if __has_builtin(__builtin_amdgcn_tanhf)
static __device__ __forceinline__ float fast_tanh(float x) { return __builtin_amdgcn_tanhf(x); }
#else
static __device__ __forceinline__ float fast_tanh(float x) {
    float r;
    // CDNA5 TRANS op; v_nop covers the 1-op trans-result hazard
    asm volatile("v_tanh_f32 %0, %1\n\tv_nop" : "=v"(r) : "v"(x));
    return r;
}
#endif

static __device__ __forceinline__ float fast_exp2(float x) { return __builtin_amdgcn_exp2f(x); }
static __device__ __forceinline__ float fast_log2(float x) { return __builtin_amdgcn_logf(x); }
static __device__ __forceinline__ float fast_rcp (float x) { return __builtin_amdgcn_rcpf(x); }
static __device__ __forceinline__ float fast_sigmoid(float x) {
    return fast_rcp(1.0f + fast_exp2(-1.4426950408889634f * x));
}

// ---------------------------------------------------------------------------
// Kernel 1: fold conv1x1 + H x H projections into one (H x 5) affine map.
// base[b,h,s] = E0[h]*st0 + E1[h]*st1 + Eld[h]*(load-demand) + Ed[h]*demand + Bb[h]
// ---------------------------------------------------------------------------
__global__ void fuse_small(const float* __restrict__ W_s,  const float* __restrict__ b_s,
                           const float* __restrict__ W_ld, const float* __restrict__ b_ld,
                           const float* __restrict__ W_d,  const float* __restrict__ b_d,
                           const float* __restrict__ W_pd, const float* __restrict__ b_pd,
                           const float* __restrict__ W_pld,const float* __restrict__ b_pld,
                           const float* __restrict__ W_pr, const float* __restrict__ b_pr,
                           const float* __restrict__ mark,
                           float* __restrict__ fused, float* __restrict__ out_mark)
{
    int h = threadIdx.x;
    if (h < H) {
        float e0 = 0.f, e1 = 0.f, eld = 0.f, ed = 0.f, bb = 0.f;
        for (int k = 0; k < H; ++k) {
            float pr  = W_pr [h * H + k];
            float pld = W_pld[h * H + k];
            float pd  = W_pd [h * H + k];
            e0  += pr  * W_s[k * 2 + 0];
            e1  += pr  * W_s[k * 2 + 1];
            eld += pld * (W_ld[k * 2 + 0] + W_ld[k * 2 + 1]);
            ed  += pd  * (W_d [k * 2 + 0] + W_d [k * 2 + 1]);
            bb  += pr * b_s[k] + pld * b_ld[k] + pd * b_d[k];
        }
        bb += b_pr[h] + b_pld[h] + b_pd[h];
        fused[0 * H + h] = e0;
        fused[1 * H + h] = e1;
        fused[2 * H + h] = eld;
        fused[3 * H + h] = ed;
        fused[4 * H + h] = bb;
    }
    if (h == 0) out_mark[0] = mark[0];
}

// ---------------------------------------------------------------------------
// Kernel 2: materialize base (B,H,S) fp32 and static_hidden^T (B,S,H) fp32.
// ---------------------------------------------------------------------------
__global__ void precompute(const float* __restrict__ statc, const float* __restrict__ dyn,
                           const float* __restrict__ W_s,   const float* __restrict__ b_s,
                           const float* __restrict__ fused,
                           float* __restrict__ base, float* __restrict__ shT)
{
    __shared__ float sE0[H], sE1[H], sEld[H], sEd[H], sBb[H], sW0[H], sW1[H], sBs[H];
    const int t = threadIdx.x;
    const int b = blockIdx.x;
    if (t < H) {
        sE0[t] = fused[t];          sE1[t] = fused[H + t];
        sEld[t] = fused[2 * H + t]; sEd[t] = fused[3 * H + t];
        sBb[t] = fused[4 * H + t];
        sW0[t] = W_s[t * 2];        sW1[t] = W_s[t * 2 + 1];
        sBs[t] = b_s[t];
    }
    __syncthreads();
    const float* st0 = statc + (size_t)b * 2 * S;
    const float* dy0 = dyn   + (size_t)b * 2 * S;
    for (int i = t; i < H * S; i += NTHREADS) {            // base: s innermost
        int h = i >> 7, s = i & (S - 1);
        float x0 = st0[s], x1 = st0[S + s];
        float ld = dy0[s], dm = dy0[S + s];
        base[((size_t)b * H + h) * S + s] =
            sE0[h] * x0 + sE1[h] * x1 + sEld[h] * (ld - dm) + sEd[h] * dm + sBb[h];
    }
    for (int i = t; i < H * S; i += NTHREADS) {            // shT: h innermost
        int s = i >> 7, h = i & (H - 1);
        float x0 = st0[s], x1 = st0[S + s];
        shT[((size_t)b * S + s) * H + h] = sW0[h] * x0 + sW1[h] * x1 + sBs[h];
    }
}

// ---------------------------------------------------------------------------
// Kernel 3: persistent decode. 128 blocks x 256 threads (8 waves), 16 batch
// rows per block, full 128-step loop in-kernel. LDS (~307KB):
//   sWp : [W_ih;W_hh]^T bf16 in WMMA fragment order (256KB)
//         index = ((kk*32 + ntile)*32 + lane)*8 + v
//   sWqp: W_pq^T bf16 in fragment order (32KB)
//   sX  : [dec | h] 16x256 bf16 (ISA A-layout compatible, 8KB)
// Wave w owns gate tiles {w, 8+w, 16+w, 24+w} -> hidden units [16w,16w+16),
// so LSTM cell state c stays in registers across all 128 steps.
// ---------------------------------------------------------------------------
__global__ __launch_bounds__(NTHREADS, 1)
void decode(const float* __restrict__ W_ih, const float* __restrict__ b_ih,
            const float* __restrict__ W_hh, const float* __restrict__ b_hh,
            const float* __restrict__ W_pq, const float* __restrict__ b_pq,
            const float* __restrict__ attn_W,
            const float* __restrict__ base, const float* __restrict__ shT,
            int* __restrict__ out_idx, float* __restrict__ out_logp)
{
    __shared__ uint32_t sWp [8 * 32 * 32 * 8];   // 256KB fragment-ordered
    __shared__ uint32_t sWqp[4 *  8 * 32 * 8];   //  32KB fragment-ordered
    __shared__ uint32_t sX  [16 * 128];          //   8KB: 16 rows x 256 bf16
    __shared__ float    sQ  [16 * 128];          //   8KB
    __shared__ float    sBias[4 * H];
    __shared__ float    sBq[H];
    __shared__ float    sAw[H];
    __shared__ int      sPtr[BTILE];

    const int tid  = threadIdx.x;
    const int wv   = tid >> 5;
    const int lane = tid & 31;
    const int b0   = blockIdx.x * BTILE;
    const int nn   = lane & 15;            // A-row m / B-col n / C-col n
    const int kh   = lane >> 4;            // K-half select
    uint16_t* sX16 = (uint16_t*)sX;

    // ---- one-time LDS fill (fragment order) -------------------------------
    // sWp[((kk*32+nt)*32+l)*8+v] = bf16pair( Wcat[n][k0], Wcat[n][k0+1] )
    //   n = nt*16 + (l&15),  k0 = kk*32 + (l>>4)*8 + 2v + (v>>2)*8,
    //   Wcat[n][k] = k<128 ? W_ih[n][k] : W_hh[n][k-128]
    for (int i = tid; i < 8 * 32 * 32 * 8; i += NTHREADS) {
        int v  = i & 7;
        int l  = (i >> 3) & 31;
        int nt = (i >> 8) & 31;
        int kk = i >> 13;
        int n  = nt * 16 + (l & 15);
        int k0 = kk * 32 + (l >> 4) * 8 + 2 * v + (v >> 2) * 8;
        float a, c;
        if (k0 < H) { a = W_ih[n * H + k0];       c = W_ih[n * H + k0 + 1]; }
        else        { a = W_hh[n * H + k0 - 128]; c = W_hh[n * H + k0 - 127]; }
        sWp[i] = (uint32_t)f2bf(a) | ((uint32_t)f2bf(c) << 16);
    }
    for (int i = tid; i < 4 * 8 * 32 * 8; i += NTHREADS) {
        int v  = i & 7;
        int l  = (i >> 3) & 31;
        int nt = (i >> 8) & 7;
        int kk = i >> 11;
        int n  = nt * 16 + (l & 15);
        int k0 = kk * 32 + (l >> 4) * 8 + 2 * v + (v >> 2) * 8;
        sWqp[i] = (uint32_t)f2bf(W_pq[n * H + k0]) |
                  ((uint32_t)f2bf(W_pq[n * H + k0 + 1]) << 16);
    }
    for (int i = tid; i < 4 * H; i += NTHREADS) sBias[i] = b_ih[i] + b_hh[i];
    if (tid < H) { sBq[tid] = b_pq[tid]; sAw[tid] = attn_W[tid]; }
    // dec0 = static_hidden[:, :, 0];  h0 = 0
    for (int i = tid; i < BTILE * H; i += NTHREADS) {
        int m = i >> 7, hh = i & (H - 1);
        sX16[m * 256 + hh]       = f2bf(shT[((size_t)(b0 + m) * S) * H + hh]);
        sX16[m * 256 + 128 + hh] = 0;
    }
    __syncthreads();

    float creg[8];
    #pragma unroll
    for (int v = 0; v < 8; ++v) creg[v] = 0.0f;

    for (int t = 0; t < S; ++t) {
        // ---- gates = [dec|h](16x256) @ [W_ih;W_hh]^T(256x512) + bias -------
        v8f acc[4];
        #pragma unroll
        for (int g = 0; g < 4; ++g) {
            float bv = sBias[g * 128 + 16 * wv + nn];
            #pragma unroll
            for (int v = 0; v < 8; ++v) acc[g][v] = bv;
        }
        #pragma unroll
        for (int kk = 0; kk < 8; ++kk) {
            // A fragment: two contiguous 16B runs (ISA 16-bit A layout)
            const uint32_t* ax = sX + nn * 128 + kk * 16 + kh * 4;
            uint4 alo = *(const uint4*)(ax);
            uint4 ahi = *(const uint4*)(ax + 8);
            v8u au = {alo.x, alo.y, alo.z, alo.w, ahi.x, ahi.y, ahi.z, ahi.w};
            v16bf a = __builtin_bit_cast(v16bf, au);
            #pragma unroll
            for (int g = 0; g < 4; ++g) {
                int nt = g * 8 + wv;                  // gate g, hidden tile wv
                v8u bu = *(const v8u*)(sWp + (((kk * 32 + nt) * 32 + lane) * 8));
                v16bf bm = __builtin_bit_cast(v16bf, bu);
                acc[g] = __builtin_amdgcn_wmma_f32_16x16x32_bf16(
                    false, a, false, bm, (short)0, acc[g], false, false);
            }
        }
        // ---- LSTM activations in registers (m = v + 8*kh, j = 16*wv + nn) --
        float hreg[8];
        #pragma unroll
        for (int v = 0; v < 8; ++v) {
            float ig = fast_sigmoid(acc[0][v]);
            float fg = fast_sigmoid(acc[1][v]);
            float gg = fast_tanh   (acc[2][v]);
            float og = fast_sigmoid(acc[3][v]);
            float cn = fg * creg[v] + ig * gg;
            creg[v]  = cn;
            hreg[v]  = og * fast_tanh(cn);
        }
        __syncthreads();                    // everyone finished reading old h
        #pragma unroll
        for (int v = 0; v < 8; ++v) {
            int m = v + 8 * kh;
            sX16[m * 256 + 128 + 16 * wv + nn] = f2bf(hreg[v]);
        }
        __syncthreads();

        // ---- q = h(16x128) @ W_pq^T(128x128) + b_pq ------------------------
        v8f qacc;
        {
            float bv = sBq[16 * wv + nn];
            #pragma unroll
            for (int v = 0; v < 8; ++v) qacc[v] = bv;
        }
        #pragma unroll
        for (int kk = 0; kk < 4; ++kk) {
            const uint32_t* ax = sX + nn * 128 + 64 + kk * 16 + kh * 4;
            uint4 alo = *(const uint4*)(ax);
            uint4 ahi = *(const uint4*)(ax + 8);
            v8u au = {alo.x, alo.y, alo.z, alo.w, ahi.x, ahi.y, ahi.z, ahi.w};
            v16bf a = __builtin_bit_cast(v16bf, au);
            v8u bu = *(const v8u*)(sWqp + (((kk * 8 + wv) * 32 + lane) * 8));
            v16bf bm = __builtin_bit_cast(v16bf, bu);
            qacc = __builtin_amdgcn_wmma_f32_16x16x32_bf16(
                false, a, false, bm, (short)0, qacc, false, false);
        }
        #pragma unroll
        for (int v = 0; v < 8; ++v) {
            int m = v + 8 * kh;
            sQ[m * 128 + 16 * wv + nn] = qacc[v];
        }
        __syncthreads();

        // ---- attention: attn[s] = sum_h aW[h]*tanh(base[b,h,s]+q[b,h]) -----
        // wave handles rows 2*wv, 2*wv+1; lane owns s = 4*lane..4*lane+3
        #pragma unroll
        for (int half = 0; half < 2; ++half) {
            int m = 2 * wv + half;
            const float* bp = base + ((size_t)(b0 + m) * H) * S + 4 * lane;
            const float* qp = sQ + m * 128;
            float a0 = 0.f, a1 = 0.f, a2 = 0.f, a3 = 0.f;
            #pragma unroll 4
            for (int hh = 0; hh < H; ++hh) {
                float4 x = *(const float4*)(bp + (size_t)hh * S);  // L2-resident
                float qv = qp[hh];
                float wt = sAw[hh];
                a0 += wt * fast_tanh(x.x + qv);
                a1 += wt * fast_tanh(x.y + qv);
                a2 += wt * fast_tanh(x.z + qv);
                a3 += wt * fast_tanh(x.w + qv);
            }
            if (lane == 0) a0 -= 10000.0f;  // mask2: s==0 suppressed
            float mvx = a0; int mix = 4 * lane;
            if (a1 > mvx) { mvx = a1; mix = 4 * lane + 1; }
            if (a2 > mvx) { mvx = a2; mix = 4 * lane + 2; }
            if (a3 > mvx) { mvx = a3; mix = 4 * lane + 3; }
            for (int off = 16; off > 0; off >>= 1) {
                float ov = __shfl_xor(mvx, off, 32);
                int   oi = __shfl_xor(mix, off, 32);
                if (ov > mvx || (ov == mvx && oi < mix)) { mvx = ov; mix = oi; }
            }
            const float L2E = 1.4426950408889634f;
            float sm = fast_exp2(L2E * (a0 - mvx)) + fast_exp2(L2E * (a1 - mvx)) +
                       fast_exp2(L2E * (a2 - mvx)) + fast_exp2(L2E * (a3 - mvx));
            for (int off = 16; off > 0; off >>= 1) sm += __shfl_xor(sm, off, 32);
            if (lane == 0) {
                out_idx [(size_t)(b0 + m) * S + t] = mix;
                out_logp[(size_t)(b0 + m) * S + t] = -0.6931471805599453f * fast_log2(sm);
                sPtr[m] = mix;
            }
        }
        __syncthreads();

        // ---- gather next decoder input: dec = static_hidden^T[b, ptr, :] ---
        for (int i = tid; i < BTILE * H; i += NTHREADS) {
            int m = i >> 7, hh = i & (H - 1);
            int p = sPtr[m];
            sX16[m * 256 + hh] = f2bf(shT[((size_t)(b0 + m) * S + p) * H + hh]);
        }
        __syncthreads();
    }
}

// ---------------------------------------------------------------------------
extern "C" void kernel_launch(void* const* d_in, const int* in_sizes, int n_in,
                              void* d_out, int out_size, void* d_ws, size_t ws_size,
                              hipStream_t stream)
{
    (void)in_sizes; (void)n_in; (void)out_size; (void)ws_size;
    const float* statc  = (const float*)d_in[0];
    const float* dyn    = (const float*)d_in[1];
    const float* mark   = (const float*)d_in[2];
    const float* W_s    = (const float*)d_in[3];
    const float* b_s    = (const float*)d_in[4];
    const float* W_ld   = (const float*)d_in[5];
    const float* b_ld   = (const float*)d_in[6];
    const float* W_d    = (const float*)d_in[7];
    const float* b_d    = (const float*)d_in[8];
    const float* W_ih   = (const float*)d_in[9];
    const float* b_ih   = (const float*)d_in[10];
    const float* W_hh   = (const float*)d_in[11];
    const float* b_hh   = (const float*)d_in[12];
    const float* W_pd   = (const float*)d_in[13];
    const float* b_pd   = (const float*)d_in[14];
    const float* W_pld  = (const float*)d_in[15];
    const float* b_pld  = (const float*)d_in[16];
    const float* W_pq   = (const float*)d_in[17];
    const float* b_pq   = (const float*)d_in[18];
    const float* W_pr   = (const float*)d_in[19];
    const float* b_pr   = (const float*)d_in[20];
    const float* attn_W = (const float*)d_in[21];

    float* fused = (float*)d_ws;                       // 5*H floats (pad to 1024)
    float* base  = fused + 1024;                       // B*H*S fp32 (L2-resident)
    float* shT   = base + (size_t)BATCH * H * S;       // B*S*H fp32

    int*   out_idx  = (int*)d_out;                               // tour_idx (B,S)
    float* out_logp = (float*)d_out + (size_t)BATCH * S;         // tour_logp (B,S)
    float* out_mark = (float*)d_out + (size_t)2 * BATCH * S;     // mark (1,)

    fuse_small<<<1, 128, 0, stream>>>(W_s, b_s, W_ld, b_ld, W_d, b_d,
                                      W_pd, b_pd, W_pld, b_pld, W_pr, b_pr,
                                      mark, fused, out_mark);
    precompute<<<BATCH, NTHREADS, 0, stream>>>(statc, dyn, W_s, b_s, fused, base, shT);
    decode<<<BATCH / BTILE, NTHREADS, 0, stream>>>(W_ih, b_ih, W_hh, b_hh,
                                                   W_pq, b_pq, attn_W,
                                                   base, shT, out_idx, out_logp);
}
